// TemporalEncoder_43525198578332
// MI455X (gfx1250) — matrix operand order repaired and checked by reading, not compiled, not run on hardware
//
#include <hip/hip_runtime.h>
#include <hip/hip_bf16.h>
#include <math.h>

#define B_   256
#define SEQ_ 512
#define PD_  64
#define H_   256
#define G4_  1024   // 4*H

#define USE_ASYNC_LDS 1   // gfx1250 global_load_async_to_lds_b128 staging

typedef __attribute__((ext_vector_type(16))) __bf16       bf16x16;
typedef __attribute__((ext_vector_type(8)))  __bf16       bf16x8;
typedef __attribute__((ext_vector_type(8)))  float        f32x8;
typedef __attribute__((ext_vector_type(4)))  unsigned int u32x4;
typedef __attribute__((ext_vector_type(8)))  unsigned int u32x8;

__device__ __forceinline__ float sigmoidf_(float x) { return 1.0f / (1.0f + __expf(-x)); }

// ---------------------------------------------------------------------------
// Pack a (K x 1024) f32 weight matrix into bf16 WMMA B-fragments.
// Tile (kb, nb): K rows [kb*32, +32), cols [nb*16, +16). Contiguous-K B layout used here:
// lane<16 -> K 0..15 (pairs in v0..7), lane>=16 -> K 16..31; N = lane&15.
// One tile = 256 dwords (32 lanes x 8).
// ---------------------------------------------------------------------------
__global__ void pack_weight_bf16(const float* __restrict__ W,
                                 unsigned* __restrict__ out, int K) {
  int idx   = blockIdx.x * blockDim.x + threadIdx.x;
  int total = (K >> 5) * 64 * 32 * 8;
  if (idx >= total) return;
  int d    = idx & 7;
  int lane = (idx >> 3) & 31;
  int nb   = (idx >> 8) & 63;
  int kb   = idx >> 14;
  int n = nb * 16 + (lane & 15);
  int k = kb * 32 + ((lane >> 4) << 4) + 2 * d;
  __bf16 lo = (__bf16)W[(size_t)k * G4_ + n];
  __bf16 hi = (__bf16)W[(size_t)(k + 1) * G4_ + n];
  unsigned short l16 = __builtin_bit_cast(unsigned short, lo);
  unsigned short h16 = __builtin_bit_cast(unsigned short, hi);
  out[idx] = (unsigned)l16 | ((unsigned)h16 << 16);
}

// pose (B,S,PD) f32 -> (S,B,PD) bf16 so each timestep slice is contiguous.
__global__ void transpose_x_bf16(const float* __restrict__ X,
                                 __bf16* __restrict__ out) {
  size_t idx   = (size_t)blockIdx.x * blockDim.x + threadIdx.x;
  size_t total = (size_t)SEQ_ * B_ * PD_;
  if (idx >= total) return;
  int k = (int)(idx % PD_);
  int b = (int)((idx / PD_) % B_);
  int t = (int)(idx / ((size_t)PD_ * B_));
  out[idx] = (__bf16)X[((size_t)b * SEQ_ + t) * PD_ + k];
}

// A fragment from row-major bf16 M; p already points at (row, khalf-base).
// A layout (matches packed B): lane<16 holds K 0..15, lane>=16 K 16..31.
__device__ __forceinline__ bf16x16 load_a_frag_at(const __bf16* __restrict__ p) {
  u32x4 lo = *(const u32x4*)(p);
  u32x4 hi = *(const u32x4*)(p + 8);
  bf16x8 alo = __builtin_bit_cast(bf16x8, lo);
  bf16x8 ahi = __builtin_bit_cast(bf16x8, hi);
  return __builtin_shufflevector(alo, ahi, 0, 1, 2, 3, 4, 5, 6, 7,
                                 8, 9, 10, 11, 12, 13, 14, 15);
}

__device__ __forceinline__ bf16x16 load_b_frag32(const unsigned* __restrict__ p) {
  return __builtin_bit_cast(bf16x16, *(const u32x8*)p);
}

// Device-wide split barrier: all 256 WGs arrive, spin on monotone counter.
__device__ __forceinline__ void grid_sync(unsigned* cnt, unsigned target) {
  __syncthreads();
  if (threadIdx.x == 0) {
    __threadfence();
    atomicAdd(cnt, 1u);
    while (__hip_atomic_load(cnt, __ATOMIC_ACQUIRE, __HIP_MEMORY_SCOPE_AGENT) < target) {
      __builtin_amdgcn_s_sleep(2);
    }
  }
  __syncthreads();
}

// ---------------------------------------------------------------------------
// One LSTM layer, persistent over SEQ timesteps.
// Grid: 256 WGs = 16 batch-blocks x 16 hidden-blocks; block = 4 wave32s,
// wave g owns the gate-g 16x16 tile of z = b + x_t*Wi + h_{t-1}*Wh.
// Weight B-fragments staged once into LDS via async global->LDS copies.
// h double-buffered: read hbuf[t&1], write hbuf[(t+1)&1].
// ---------------------------------------------------------------------------
template <int KBX, bool STAGE_WI, bool WRITE_YS, bool WRITE_OUT>
__global__ __launch_bounds__(128) void lstm_layer_wmma(
    const __bf16* __restrict__ X,        // (SEQ,B,Kx)
    const unsigned* __restrict__ wiPack, // (KBX,64) tiles
    const unsigned* __restrict__ whPack, // (8,64) tiles
    const float* __restrict__ bias,      // (4H)
    float* __restrict__ c,               // (B,H) f32
    __bf16* __restrict__ hA,             // (B,H)
    __bf16* __restrict__ hB,             // (B,H)
    __bf16* __restrict__ ysOut,          // (SEQ,B,H) | unused
    float* __restrict__ finalOut,        // (B,H)     | unused
    unsigned* __restrict__ cnt) {
  constexpr int KB_LDS = (STAGE_WI ? KBX + 8 : 8);         // K-blocks per gate in LDS
  __shared__ __align__(16) unsigned wlds[KB_LDS * 4 * 256]; // <= 40 KB
  __shared__ float zs[4 * 16 * 16];

  const int wg   = blockIdx.x;
  const int b0   = (wg >> 4) << 4;     // batch block start
  const int jb   = wg & 15;            // hidden block
  const int wave = threadIdx.x >> 5;   // gate 0..3 (i,f,g,o)
  const int lane = threadIdx.x & 31;
  const unsigned nwg = gridDim.x;

  // ---- stage this WG's weight B-fragments into LDS (once) ----
  {
    const int chunks = KB_LDS * 4 * 64;            // 16B chunks
    for (int i = threadIdx.x; i < chunks; i += 128) {
      int tile = i >> 6;                           // local tile id
      int gate = tile / KB_LDS;
      int kbi  = tile - gate * KB_LDS;
      int nbG  = gate * 16 + jb;
      const unsigned* src;
      if (STAGE_WI && kbi < KBX)
        src = wiPack + ((size_t)(kbi * 64 + nbG) << 8);
      else
        src = whPack + ((size_t)((kbi - (STAGE_WI ? KBX : 0)) * 64 + nbG) << 8);
      src += (i & 63) << 2;                        // dword offset in tile
#if USE_ASYNC_LDS
      unsigned ldsoff = (unsigned)(uintptr_t)(&wlds[(size_t)i << 2]);
      asm volatile("global_load_async_to_lds_b128 %0, %1, off"
                   :: "v"(ldsoff), "v"(src) : "memory");
#else
      *(u32x4*)&wlds[(size_t)i << 2] = *(const u32x4*)src;
#endif
    }
#if USE_ASYNC_LDS
    asm volatile("s_wait_asynccnt 0x0" ::: "memory");
#endif
    __syncthreads();
  }

  // ---- hoisted per-lane addressing ----
  const int arow  = lane & 15;
  const int ahalf = (lane >> 4) << 4;                       // 0 or 16
  const __bf16* hA_l = hA + (size_t)(b0 + arow) * H_ + ahalf;
  const __bf16* hB_l = hB + (size_t)(b0 + arow) * H_ + ahalf;
  const __bf16* x_l  = X + (size_t)(b0 + arow) * KBX * 32 + ahalf;
  const unsigned* wl = &wlds[((wave * KB_LDS) << 8) + (lane << 3)];
  const unsigned* wiG = wiPack + (((size_t)(wave * 16 + jb) * 32 + lane) << 3);
  const float bcol = bias[(wave * 16 + jb) * 16 + (lane & 15)];
  constexpr size_t XSTEP = (size_t)B_ * KBX * 32;

  for (int t = 0; t < SEQ_; ++t) {
    const __bf16* hr_l = (t & 1) ? hB_l : hA_l;
    __bf16*       hw   = (t & 1) ? hA : hB;
    const __bf16* xt_l = x_l + (size_t)t * XSTEP;
    if (t + 1 < SEQ_) __builtin_prefetch(xt_l + XSTEP, 0, 3);

    // ---- z tile = bias + x_t*Wi + h_{t-1}*Wh ----
    f32x8 acc = {bcol, bcol, bcol, bcol, bcol, bcol, bcol, bcol};
#pragma unroll
    for (int kb = 0; kb < KBX; ++kb) {
      bf16x16 a  = load_a_frag_at(xt_l + kb * 32);
      bf16x16 bm = STAGE_WI ? load_b_frag32(wl + (kb << 8))
                            : load_b_frag32(wiG + ((size_t)(kb * 64) << 8));
      acc = __builtin_amdgcn_wmma_f32_16x16x32_bf16(false, a, false, bm,
                                                    (short)0, acc, false, false);
    }
    const unsigned* wh_l = wl + ((STAGE_WI ? KBX : 0) << 8);
#pragma unroll
    for (int kb = 0; kb < 8; ++kb) {
      bf16x16 a  = load_a_frag_at(hr_l + kb * 32);
      bf16x16 bm = load_b_frag32(wh_l + (kb << 8));
      acc = __builtin_amdgcn_wmma_f32_16x16x32_bf16(false, a, false, bm,
                                                    (short)0, acc, false, false);
    }

    // ---- spill tile to LDS (C/D layout: VGPR r -> M=r(+8), N=lane&15) ----
    {
      int mbase = (lane >> 4) << 3;
      int n     = lane & 15;
#pragma unroll
      for (int r = 0; r < 8; ++r)
        zs[wave * 256 + (mbase + r) * 16 + n] = acc[r];
    }
    __syncthreads();

    // ---- fused gate math: 256 (b,j) pairs, 2 per thread ----
#pragma unroll
    for (int e = 0; e < 2; ++e) {
      int   idx = (threadIdx.x << 1) + e;
      int   m = idx >> 4, n = idx & 15;
      int   bi = b0 + m, jj = (jb << 4) + n;
      float ig = sigmoidf_(zs[idx]);
      float fg = sigmoidf_(zs[256 + idx]);
      float gg = tanhf(zs[512 + idx]);
      float og = sigmoidf_(zs[768 + idx]);
      size_t sidx = (size_t)bi * H_ + jj;
      float cv = fg * c[sidx] + ig * gg;
      float hv = og * tanhf(cv);
      c[sidx]  = cv;
      hw[sidx] = (__bf16)hv;
      if (WRITE_YS) ysOut[((size_t)t * B_ + bi) * H_ + jj] = (__bf16)hv;
      if (WRITE_OUT && t == SEQ_ - 1) finalOut[sidx] = hv;
    }

    grid_sync(cnt, (unsigned)(t + 1) * nwg);
  }
}

extern "C" void kernel_launch(void* const* d_in, const int* in_sizes, int n_in,
                              void* d_out, int out_size, void* d_ws, size_t ws_size,
                              hipStream_t stream) {
  (void)in_sizes; (void)n_in; (void)out_size; (void)ws_size;
  const float* pose = (const float*)d_in[0];
  const float* Wi0  = (const float*)d_in[1];
  const float* Wh0  = (const float*)d_in[2];
  const float* b0   = (const float*)d_in[3];
  const float* Wi1  = (const float*)d_in[4];
  const float* Wh1  = (const float*)d_in[5];
  const float* b1   = (const float*)d_in[6];
  float* out = (float*)d_out;

  char*  ws  = (char*)d_ws;
  size_t off = 0;
  auto alloc = [&](size_t bytes) -> void* {
    void* p = ws + off;
    off += (bytes + 255) & ~(size_t)255;
    return p;
  };

  unsigned* cnt0  = (unsigned*)alloc(256);
  unsigned* cnt1  = (unsigned*)alloc(256);
  unsigned* wi0P  = (unsigned*)alloc((size_t)PD_ * G4_ * 2);
  unsigned* wh0P  = (unsigned*)alloc((size_t)H_ * G4_ * 2);
  unsigned* wi1P  = (unsigned*)alloc((size_t)H_ * G4_ * 2);
  unsigned* wh1P  = (unsigned*)alloc((size_t)H_ * G4_ * 2);
  float*    cbuf  = (float*)alloc((size_t)B_ * H_ * 4);
  __bf16*   hAbuf = (__bf16*)alloc((size_t)B_ * H_ * 2);
  __bf16*   hBbuf = (__bf16*)alloc((size_t)B_ * H_ * 2);
  __bf16*   xbf   = (__bf16*)alloc((size_t)SEQ_ * B_ * PD_ * 2);  // 16 MB
  __bf16*   ys0   = (__bf16*)alloc((size_t)SEQ_ * B_ * H_ * 2);   // 64 MB

  hipMemsetAsync(cnt0, 0, 512, stream);                 // cnt0 + cnt1
  hipMemsetAsync(cbuf, 0, (size_t)B_ * H_ * 4, stream);
  hipMemsetAsync(hAbuf, 0, (size_t)B_ * H_ * 2, stream);

  {
    int n0 = (PD_ >> 5) * 64 * 32 * 8;
    int n1 = (H_ >> 5) * 64 * 32 * 8;
    pack_weight_bf16<<<(n0 + 255) / 256, 256, 0, stream>>>(Wi0, wi0P, PD_);
    pack_weight_bf16<<<(n1 + 255) / 256, 256, 0, stream>>>(Wh0, wh0P, H_);
    pack_weight_bf16<<<(n1 + 255) / 256, 256, 0, stream>>>(Wi1, wi1P, H_);
    pack_weight_bf16<<<(n1 + 255) / 256, 256, 0, stream>>>(Wh1, wh1P, H_);
    size_t nx = (size_t)SEQ_ * B_ * PD_;
    transpose_x_bf16<<<(unsigned)((nx + 255) / 256), 256, 0, stream>>>(pose, xbf);
  }

  // layer 0: K=64 (KBX=2), stage Wi+Wh in LDS, emit ys0
  lstm_layer_wmma<2, true, true, false><<<256, 128, 0, stream>>>(
      xbf, wi0P, wh0P, b0, cbuf, hAbuf, hBbuf, ys0, nullptr, cnt0);
  // layer 1: K=256 (KBX=8), stage Wh in LDS (Wi stays L2-resident), write d_out
  lstm_layer_wmma<8, false, false, true><<<256, 128, 0, stream>>>(
      ys0, wi1P, wh1P, b1, cbuf, hAbuf, hBbuf, nullptr, out, cnt1);
}
